// BahdanauAttention_5506148073830
// MI455X (gfx1250) — matrix operand order repaired and checked by reading, not compiled, not run on hardware
//
#include <hip/hip_runtime.h>
#include <hip/hip_bf16.h>
#include <math.h>

// Problem constants (match reference)
#define B_  32
#define S_  4096
#define D_  512
#define U_  512

// LDS strides (in _Float16 elements). Both are multiples of 8 (16B-aligned rows
// for b128 reads) with row-to-row bank phase that spreads 16 lanes across all
// 64 LDS banks:
//   VAL_STRIDE 520: 1040B/row -> (1040/4)%64 = 4  banks apart per row
//   W1_STRIDE  536: 1072B/row -> (1072/4)%64 = 12 banks apart per row
#define VAL_STRIDE 520
#define W1_STRIDE  536

typedef __attribute__((ext_vector_type(16))) _Float16 v16h;
typedef __attribute__((ext_vector_type(8)))  _Float16 v8h;
typedef __attribute__((ext_vector_type(4)))  _Float16 v4h;
typedef __attribute__((ext_vector_type(8)))  float    v8f;

// ---------------------------------------------------------------------------
// Kernel 1: proj_q[b,u] = sum_k query[b,k] * W2[k,u] + b2[u]     (tiny GEMM)
// ---------------------------------------------------------------------------
__global__ __launch_bounds__(256)
void projq_kernel(const float* __restrict__ query,
                  const float* __restrict__ W2,
                  const float* __restrict__ b2,
                  float* __restrict__ pq)
{
    const int idx = blockIdx.x * 256 + threadIdx.x;     // [0, B*U)
    if (idx >= B_ * U_) return;
    const int b = idx >> 9;         // /512
    const int u = idx & (U_ - 1);
    const float* qr = query + b * D_;
    float acc = b2[u];
    #pragma unroll 8
    for (int k = 0; k < D_; ++k)
        acc = fmaf(qr[k], W2[k * U_ + u], acc);
    pq[idx] = acc;
}

// ---------------------------------------------------------------------------
// Kernel 2 (WMMA core): fused  score[b,s] = tanh(values@W1 + b1 + pq) . V + bv
//
// Block: 256 threads = 8 waves; each wave owns a 16-row s-tile, block = 128 rows.
// values tile (128 x 512) is converted f32->f16 into LDS ONCE per block, then
// reused for all 32 N-tiles (2 N-tiles per pass, sharing the A fragment).
// Inner loop = 3x ds_load_b128 per v_wmma_f32_16x16x32_f16.
// Grid: (S/128, B)
// ---------------------------------------------------------------------------
__global__ __launch_bounds__(256)
void score_kernel(const float* __restrict__ values,
                  const float* __restrict__ W1,
                  const float* __restrict__ b1,
                  const float* __restrict__ pq,
                  const float* __restrict__ V,
                  const float* __restrict__ bv,
                  float* __restrict__ scores)
{
    __shared__ __align__(16) _Float16 vlds[128 * VAL_STRIDE];  // 130.0 KB
    __shared__ __align__(16) _Float16 w1s[32 * W1_STRIDE];     //  33.5 KB

    const int b    = blockIdx.y;
    const int s0   = blockIdx.x * 128;
    const int tid  = threadIdx.x;
    const int wave = tid >> 5;
    const int lane = tid & 31;
    const int m    = lane & 15;               // A row / B,C column within tile
    const int kbA  = (lane < 16) ? 0 : 16;    // A half-wave K base (contiguous order used here, K = 16h + e)
    const int kbB  = (lane < 16) ? 0 : 16;    // B uses the same order as A: halves hold K 0..15 / 16..31

    // ---- stage values[b, s0:s0+128, :] into LDS as f16 (once per block) ----
    {
        const float4* vsrc = (const float4*)(values + ((size_t)(b * S_ + s0)) * D_);
        for (int i = tid; i < 128 * (D_ / 4); i += 256) {
            const int r = i >> 7;            // row 0..127
            const int c = i & 127;           // float4 index 0..127
            const float4 v = vsrc[r * (D_ / 4) + c];
            v4h h;
            h[0] = (_Float16)v.x; h[1] = (_Float16)v.y;
            h[2] = (_Float16)v.z; h[3] = (_Float16)v.w;
            *(v4h*)&vlds[r * VAL_STRIDE + c * 4] = h;
        }
    }

    float sacc[8];
    #pragma unroll
    for (int r = 0; r < 8; ++r) sacc[r] = 0.0f;

    const float bvv = bv[0];
    const _Float16* arow = &vlds[(wave * 16 + m) * VAL_STRIDE];

    for (int nt = 0; nt < U_ / 32; ++nt) {
        const int n0 = nt * 32;

        // ---- stage W1[:, n0:n0+32] into LDS (transposed, f16) ----
        __syncthreads();
        for (int i = tid; i < (32 * D_) / 4; i += 256) {
            const int k = i >> 3;
            const int c = (i & 7) * 4;
            const float4 v = *(const float4*)(W1 + (size_t)k * U_ + n0 + c);
            w1s[(c + 0) * W1_STRIDE + k] = (_Float16)v.x;
            w1s[(c + 1) * W1_STRIDE + k] = (_Float16)v.y;
            w1s[(c + 2) * W1_STRIDE + k] = (_Float16)v.z;
            w1s[(c + 3) * W1_STRIDE + k] = (_Float16)v.w;
        }
        __syncthreads();

        // prefetch next W1 column-tile into cache (global_prefetch_b8)
        if (nt + 1 < U_ / 32) {
            const float* nxt = W1 + n0 + 32;
            for (int k = tid; k < D_; k += 256)
                __builtin_prefetch(nxt + (size_t)k * U_, 0, 1);
        }

        v8f acc0 = {};
        v8f acc1 = {};
        #pragma unroll 4
        for (int kt = 0; kt < D_ / 32; ++kt) {
            const int k0 = kt * 32;

            // ---- A fragment (shared by both N-sub-tiles), from LDS ----
            // lane<16 : e=0..15 -> K=k0+e
            // lane>=16: e=0..15 -> K=k0+16+e
            const v8h a0 = *(const v8h*)(arow + k0 + kbA);
            const v8h a1 = *(const v8h*)(arow + k0 + kbA + 8);
            v16h af;
            #pragma unroll
            for (int e = 0; e < 8; ++e) { af[e] = a0[e]; af[8 + e] = a1[e]; }

            // ---- B fragments: W1 local columns m and 16+m ----
            const v8h* bp0 = (const v8h*)&w1s[m * W1_STRIDE + k0 + kbB];
            const v8h* bp1 = (const v8h*)&w1s[(16 + m) * W1_STRIDE + k0 + kbB];
            const v8h b00 = bp0[0], b01 = bp0[1];
            const v8h b10 = bp1[0], b11 = bp1[1];
            v16h bf0, bf1;
            #pragma unroll
            for (int e = 0; e < 8; ++e) {
                bf0[e] = b00[e]; bf0[8 + e] = b01[e];
                bf1[e] = b10[e]; bf1[8 + e] = b11[e];
            }

            acc0 = __builtin_amdgcn_wmma_f32_16x16x32_f16(
                false, af, false, bf0, (short)0, acc0, false, false);
            acc1 = __builtin_amdgcn_wmma_f32_16x16x32_f16(
                false, af, false, bf1, (short)0, acc1, false, false);
        }

        // ---- fused epilogue: tanh(acc + b1[u] + pq[b,u]) * V[u] ----
        // C/D layout: lanes 0-15 hold rows 0..7 (VGPR r -> M=r), lanes 16-31 rows 8..15.
        const int   u0   = n0 + m;
        const int   u1   = n0 + 16 + m;
        const float b1u0 = b1[u0],            b1u1 = b1[u1];
        const float pq0  = pq[b * U_ + u0],   pq1  = pq[b * U_ + u1];
        const float vu0  = V[u0],             vu1  = V[u1];
        #pragma unroll
        for (int r = 0; r < 8; ++r) {
            const float t0 = tanhf(acc0[r] + b1u0 + pq0);
            const float t1 = tanhf(acc1[r] + b1u1 + pq1);
            sacc[r] = fmaf(t0, vu0, sacc[r]);
            sacc[r] = fmaf(t1, vu1, sacc[r]);
        }
    }

    // ---- reduce over the 16 lanes of each half-wave (wave32 shuffles) ----
    #pragma unroll
    for (int r = 0; r < 8; ++r) {
        float v = sacc[r];
        v += __shfl_xor(v, 1, 32);
        v += __shfl_xor(v, 2, 32);
        v += __shfl_xor(v, 4, 32);
        v += __shfl_xor(v, 8, 32);
        sacc[r] = v;
    }
    if (lane == 0 || lane == 16) {
        const int mbase = (lane == 0) ? 0 : 8;
        #pragma unroll
        for (int r = 0; r < 8; ++r)
            scores[(size_t)b * S_ + s0 + wave * 16 + mbase + r] = sacc[r] + bvv;
    }
}

// ---------------------------------------------------------------------------
// Kernel 3: softmax over S per batch.  Grid: (B), block 256.
// ---------------------------------------------------------------------------
__global__ __launch_bounds__(256)
void softmax_kernel(const float* __restrict__ scores,
                    float* __restrict__ weights)
{
    __shared__ float red[256];
    const int b   = blockIdx.x;
    const int tid = threadIdx.x;
    const float* sr = scores + (size_t)b * S_;

    float lmax = -INFINITY;
    for (int s = tid; s < S_; s += 256) lmax = fmaxf(lmax, sr[s]);
    red[tid] = lmax; __syncthreads();
    for (int off = 128; off > 0; off >>= 1) {
        if (tid < off) red[tid] = fmaxf(red[tid], red[tid + off]);
        __syncthreads();
    }
    const float gmax = red[0];
    __syncthreads();

    float lsum = 0.0f;
    for (int s = tid; s < S_; s += 256) lsum += __expf(sr[s] - gmax);
    red[tid] = lsum; __syncthreads();
    for (int off = 128; off > 0; off >>= 1) {
        if (tid < off) red[tid] += red[tid + off];
        __syncthreads();
    }
    const float inv = 1.0f / red[0];

    for (int s = tid; s < S_; s += 256)
        weights[(size_t)b * S_ + s] = __expf(sr[s] - gmax) * inv;
}

// ---------------------------------------------------------------------------
// Kernel 4: partial context sums over S-chunks (deterministic, no atomics).
// Grid: (D/256, NCHUNK, B), block 256.  partial[(chunk*B + b)*D + d]
// ---------------------------------------------------------------------------
#define NCHUNK 8
__global__ __launch_bounds__(256)
void context_partial_kernel(const float* __restrict__ values,
                            const float* __restrict__ weights,
                            float* __restrict__ partial)
{
    const int d     = blockIdx.x * 256 + threadIdx.x;
    const int chunk = blockIdx.y;
    const int b     = blockIdx.z;
    const int sBeg  = chunk * (S_ / NCHUNK);

    const float* w  = weights + (size_t)b * S_ + sBeg;
    const float* vp = values + ((size_t)b * S_ + sBeg) * D_ + d;

    float acc = 0.0f;
    #pragma unroll 4
    for (int s = 0; s < S_ / NCHUNK; ++s)
        acc = fmaf(w[s], vp[(size_t)s * D_], acc);

    partial[((size_t)chunk * B_ + b) * D_ + d] = acc;
}

// ---------------------------------------------------------------------------
// Kernel 5: reduce chunk partials -> context[b,d]
// ---------------------------------------------------------------------------
__global__ __launch_bounds__(256)
void context_reduce_kernel(const float* __restrict__ partial,
                           float* __restrict__ context)
{
    const int idx = blockIdx.x * 256 + threadIdx.x;   // b*D + d
    if (idx >= B_ * D_) return;
    float acc = 0.0f;
    #pragma unroll
    for (int c = 0; c < NCHUNK; ++c)
        acc += partial[(size_t)c * B_ * D_ + idx];
    context[idx] = acc;
}

// ---------------------------------------------------------------------------
extern "C" void kernel_launch(void* const* d_in, const int* in_sizes, int n_in,
                              void* d_out, int out_size, void* d_ws, size_t ws_size,
                              hipStream_t stream)
{
    (void)in_sizes; (void)n_in; (void)out_size; (void)ws_size;

    const float* query  = (const float*)d_in[0];   // [B, D]
    const float* values = (const float*)d_in[1];   // [B, S, D]
    const float* W1     = (const float*)d_in[2];   // [D, U]
    const float* b1     = (const float*)d_in[3];   // [U]
    const float* W2     = (const float*)d_in[4];   // [D, U]
    const float* b2     = (const float*)d_in[5];   // [U]
    const float* V      = (const float*)d_in[6];   // [U, 1]
    const float* bv     = (const float*)d_in[7];   // [1]

    float* out      = (float*)d_out;
    float* context  = out;                 // [B, D]  (first output)
    float* weights  = out + B_ * D_;       // [B, S, 1] (second output)

    float* ws       = (float*)d_ws;
    float* pq       = ws;                               // B*U floats
    float* scores   = ws + B_ * U_;                     // B*S floats
    float* partial  = ws + B_ * U_ + (size_t)B_ * S_;   // NCHUNK*B*D floats

    // 1) proj_q = query @ W2 + b2
    projq_kernel<<<(B_ * U_ + 255) / 256, 256, 0, stream>>>(query, W2, b2, pq);

    // 2) fused WMMA: scores = tanh(values@W1 + b1 + proj_q) . V + bv
    dim3 gScore(S_ / 128, B_);
    score_kernel<<<gScore, 256, 0, stream>>>(values, W1, b1, pq, V, bv, scores);

    // 3) softmax over S
    softmax_kernel<<<B_, 256, 0, stream>>>(scores, weights);

    // 4) context partials over S-chunks
    dim3 gPart(D_ / 256, NCHUNK, B_);
    context_partial_kernel<<<gPart, 256, 0, stream>>>(values, weights, partial);

    // 5) reduce partials
    context_reduce_kernel<<<(B_ * D_ + 255) / 256, 256, 0, stream>>>(partial, context);
}
